// Memory_26362509263544
// MI455X (gfx1250) — matrix operand order repaired and checked by reading, not compiled, not run on hardware
//
#include <hip/hip_runtime.h>

typedef __attribute__((ext_vector_type(2)))  float          v2f;
typedef __attribute__((ext_vector_type(4)))  float          v4f;
typedef __attribute__((ext_vector_type(8)))  float          v8f;
typedef __attribute__((ext_vector_type(16))) __bf16         v16bf;
typedef __attribute__((ext_vector_type(16))) unsigned short v16us;
typedef __attribute__((ext_vector_type(4)))  unsigned int   u32x4;
typedef __attribute__((ext_vector_type(8)))  int            i32x8;
typedef __attribute__((ext_vector_type(4)))  int            i32x4;

#define B_   4
#define DE   128
#define DO   512
#define KK   12960        // T*H*W = 8*30*54   (= 810 * 16, exact)
#define QQ   1620         // H*W   = 30*54     (= 101*16 + 4)
#define QT   102          // ceil(QQ/16)
#define SCALE 0.08838834764831845f   // 1/sqrt(128)

// round-to-nearest-even float -> bf16 bits
static __device__ inline unsigned short f2bf(float f) {
  unsigned u = __builtin_bit_cast(unsigned, f);
  unsigned r = u + 0x7FFFu + ((u >> 16) & 1u);
  return (unsigned short)(r >> 16);
}

// Issue a TDM 2D tile load (global -> LDS), 4-byte elements.
// D# built per CDNA5 ISA §8.3/8.4. Tensor dims set huge: all reads are
// in-bounds by construction (tile origins clamped), so OOB logic is unused.
static __device__ inline void tdm_load_2d(unsigned lds_addr, const void* gptr,
                                          unsigned tile_x, unsigned tile_y,
                                          unsigned stride0_elems) {
  unsigned long long ga = (unsigned long long)(uintptr_t)gptr;
  const unsigned TD0 = 1u << 20, TD1 = 1u << 20;
  u32x4 g0;
  g0[0] = 1u;                                             // count=1 (valid), user mode
  g0[1] = lds_addr;                                       // LDS byte address
  g0[2] = (unsigned)(ga & 0xFFFFFFFFu);                   // global_addr[31:0]
  g0[3] = (unsigned)((ga >> 32) & 0x1FFFFFFu) | (2u << 30); // global_addr[56:32] | type=2
  i32x8 g1;
  g1[0] = (int)(2u << 16);                                // workgroup_mask=0, data_size=2 (4B)
  g1[1] = (int)((TD0 & 0xFFFFu) << 16);                   // tensor_dim0[15:0]
  g1[2] = (int)((TD0 >> 16) | ((TD1 & 0xFFFFu) << 16));   // tensor_dim0[31:16] | tensor_dim1[15:0]
  g1[3] = (int)((TD1 >> 16) | (tile_x << 16));            // tensor_dim1[31:16] | tile_dim0
  g1[4] = (int)(tile_y & 0xFFFFu);                        // tile_dim1 | tile_dim2=0
  g1[5] = (int)stride0_elems;                             // tensor_dim0_stride[31:0]
  g1[6] = 0;                                              // stride0[47:32] | stride1[15:0]
  g1[7] = 0;                                              // stride1[47:16]
  i32x4 z4 = {};
  i32x8 z8 = {};
  __builtin_amdgcn_tensor_load_to_lds(g0, g1, z4, z4, z8, 0);
}

// -------- Kernel 1: logits[b,k,q] = (1/sqrt(De)) * sum_d m_in[b,d,k]*q_in[b,d,q]
// fp32 WMMA 16x16x4. q_in tile (16q x 128d) TDM-loaded to LDS once per block.
__global__ void gemm1_logits(const float* __restrict__ m_in,
                             const float* __restrict__ q_in,
                             float* __restrict__ logits) {
  __shared__ float sq[DE * 16];          // [d][q] : 8 KB

  const int lane  = threadIdx.x & 31;
  const int wave  = threadIdx.x >> 5;
  const int b     = blockIdx.z;
  const int qb    = min((int)blockIdx.y * 16, QQ - 16);        // clamped tile origin
  const int ktile = min((int)blockIdx.x * 8 + wave, KK / 16 - 1);
  const int hi    = lane >> 4;
  const int lo    = lane & 15;

  const float* qi = q_in + (size_t)b * DE * QQ + qb;
  if (wave == 0) {
    tdm_load_2d((unsigned)(uintptr_t)sq, qi, /*x=*/16, /*y=*/DE, /*stride=*/QQ);
    __builtin_amdgcn_s_wait_tensorcnt(0);
  }
  __syncthreads();

  const float* mi = m_in + (size_t)b * DE * KK;
  const int k = ktile * 16 + lo;         // A row (M), always < KK

  v8f c = {};
  for (int d0 = 0; d0 < DE; d0 += 4) {
    const int d = d0 + hi * 2;           // this lane's two K(=d) slots
    v2f a, bb;
    a.x  = mi[(size_t)d       * KK + k];
    a.y  = mi[(size_t)(d + 1) * KK + k];
    bb.x = sq[d * 16 + lo];
    bb.y = sq[(d + 1) * 16 + lo];
    c = __builtin_amdgcn_wmma_f32_16x16x4_f32(false, a, false, bb,
                                              (short)0, c, false, false);
  }

  // C/D layout: VGPR r -> M = r + hi*8, N = lo. Unconditional stores:
  // clamped tiles recompute identical values -> benign duplicate writes.
#pragma unroll
  for (int r = 0; r < 8; ++r) {
    const int kk = ktile * 16 + r + hi * 8;
    logits[((size_t)b * KK + kk) * QQ + qb + lo] = c[r] * SCALE;
  }
}

// -------- Kernel 2: in-place online softmax over k (plus const row, dropped)
__global__ void softmax_k(float* __restrict__ p, const float* __restrict__ cst) {
  const int q = blockIdx.x * blockDim.x + threadIdx.x;
  const int b = blockIdx.y;
  if (q >= QQ) return;
  float* col = p + (size_t)b * KK * QQ + q;

  const float cval = cst[0];
  float m = cval, s = 1.0f;
  for (int k = 0; k < KK; ++k) {
    float v = col[(size_t)k * QQ];
    if (v > m) { s = s * __expf(m - v) + 1.0f; m = v; }
    else       { s += __expf(v - m); }
  }
  const float inv = 1.0f / s;
  for (int k = 0; k < KK; ++k) {
    float v = col[(size_t)k * QQ];
    col[(size_t)k * QQ] = __expf(v - m) * inv;
  }
}

// -------- Kernel 3: mem[b,do,q] = sum_k m_out[b,do,k] * p[b,k,q]
// bf16 WMMA 16x16x32; p tile (32k x 16q) TDM-loaded to LDS per k-step,
// m_out rows read as b128 and converted to bf16 in-register.
__global__ void gemm2_mem(const float* __restrict__ m_out,
                          const float* __restrict__ p,
                          float* __restrict__ mem) {
  __shared__ float sp[32 * 16];          // [k][q] : 2 KB

  const int lane   = threadIdx.x & 31;
  const int wave   = threadIdx.x >> 5;
  const int b      = blockIdx.z;
  const int qb     = min((int)blockIdx.y * 16, QQ - 16);   // clamped tile origin
  const int dotile = blockIdx.x * 8 + wave;                // 0..31, exact
  const int hi     = lane >> 4;
  const int lo     = lane & 15;

  const float* arow = m_out + ((size_t)b * DO + dotile * 16 + lo) * KK;
  const float* pb   = p + (size_t)b * KK * QQ + qb;
  const unsigned lds = (unsigned)(uintptr_t)sp;

  v8f c = {};
  for (int k0 = 0; k0 < KK; k0 += 32) {
    __syncthreads();                      // previous tile fully consumed
    if (wave == 0) {
      tdm_load_2d(lds, pb + (size_t)k0 * QQ, /*x=*/16, /*y=*/32, /*stride=*/QQ);
      __builtin_amdgcn_s_wait_tensorcnt(0);
    }
    if (k0 + 128 < KK)
      __builtin_prefetch(arow + k0 + 128, 0, 1);
    __syncthreads();

    // A fragment: bf16 16x32. Lane holds M = lo; halves j:
    //   K = (j<8 ? hi*8 + j : 16 + hi*8 + (j-8))  -> two aligned 32B runs.
    const v4f* r1 = (const v4f*)(arow + k0 + hi * 8);
    const v4f* r2 = (const v4f*)(arow + k0 + 16 + hi * 8);
    v4f a0 = r1[0], a1 = r1[1], a2 = r2[0], a3 = r2[1];
    v16us aus;
#pragma unroll
    for (int j = 0; j < 4; ++j) {
      aus[j]      = f2bf(a0[j]);
      aus[j + 4]  = f2bf(a1[j]);
      aus[j + 8]  = f2bf(a2[j]);
      aus[j + 12] = f2bf(a3[j]);
    }
    // B fragment: bf16 32x16. Lane holds N = lo; half j: K = (j<8 ? hi*8 + j : 16 + hi*8 + (j-8)).
    v16us bus;
#pragma unroll
    for (int j = 0; j < 16; ++j)
      bus[j] = f2bf(sp[(hi * 8 + j + (j >= 8 ? 8 : 0)) * 16 + lo]);

    c = __builtin_amdgcn_wmma_f32_16x16x32_bf16(
        false, __builtin_bit_cast(v16bf, aus),
        false, __builtin_bit_cast(v16bf, bus),
        (short)0, c, false, false);
  }

  // C/D layout: VGPR r -> M = r + hi*8, N = lo. Unconditional stores
  // (clamped q tiles duplicate identical values).
#pragma unroll
  for (int r = 0; r < 8; ++r) {
    const int d = dotile * 16 + r + hi * 8;
    mem[((size_t)b * DO + d) * QQ + qb + lo] = c[r];
  }
}

extern "C" void kernel_launch(void* const* d_in, const int* in_sizes, int n_in,
                              void* d_out, int out_size, void* d_ws, size_t ws_size,
                              hipStream_t stream) {
  (void)in_sizes; (void)n_in; (void)out_size; (void)d_ws; (void)ws_size;
  const float* m_in  = (const float*)d_in[0];
  const float* m_out = (const float*)d_in[1];
  const float* q_in  = (const float*)d_in[2];
  const float* cst   = (const float*)d_in[3];

  float* out = (float*)d_out;
  float* mem = out;                            // B*DO*QQ floats
  float* p   = out + (size_t)B_ * DO * QQ;     // B*KK*QQ floats (logits -> p in place)

  dim3 g1((KK / 16 + 7) / 8, QT, B_);          // 810 k-tiles, 8 per block
  gemm1_logits<<<g1, 256, 0, stream>>>(m_in, q_in, p);

  dim3 g2((QQ + 255) / 256, B_);
  softmax_k<<<g2, 256, 0, stream>>>(p, cst);

  dim3 g3(DO / 128, QT, B_);                   // 32 do-tiles, 8 per block
  gemm2_mem<<<g3, 256, 0, stream>>>(m_out, p, mem);
}
